// SpatialSelfConv_66434554135038
// MI455X (gfx1250) — matrix operand order repaired and checked by reading, not compiled, not run on hardware
//
#include <hip/hip_runtime.h>
#include <hip/hip_bf16.h>
#include <math.h>

typedef __attribute__((ext_vector_type(16))) __bf16 v16bf;
typedef __attribute__((ext_vector_type(8)))  __bf16 v8bf;
typedef __attribute__((ext_vector_type(8)))  float  v8f;
typedef __attribute__((ext_vector_type(4)))  unsigned int u32x4;
typedef __attribute__((ext_vector_type(4)))  int i32x4;
typedef __attribute__((ext_vector_type(8)))  int i32x8;

#define HH 56
#define WW 56
#define HW 3136           // H*W
#define ROWS 6144         // B*C
#define HID 512
#define KK 49
#define NPAD 64
#define LN_EPS 1e-5f

// ---------------------------------------------------------------------------
// combine two contiguous 16-byte halves into one WMMA operand register set
// ---------------------------------------------------------------------------
__device__ __forceinline__ v16bf make_frag(v8bf lo, v8bf hi) {
    v16bf r;
#pragma unroll
    for (int i = 0; i < 8; ++i) { r[i] = lo[i]; r[i + 8] = hi[i]; }
    return r;
}

// ---------------------------------------------------------------------------
// Tensor Data Mover: 2D tile load global -> LDS  (D# per ISA 08_async_tensor)
//   group0: count=1 | lds_addr | global_addr | type=2
//   group1: data_size | tensor_dim0/1 | tile_dim0/1 | tensor_dim0_stride
// ---------------------------------------------------------------------------
__device__ __forceinline__ void tdm_load_2d(unsigned lds_addr, const void* gptr,
                                            unsigned tensor_d0, unsigned tensor_d1,
                                            unsigned tile_d0, unsigned tile_d1,
                                            unsigned stride_d0, unsigned dsize_code) {
    unsigned long long ga = (unsigned long long)gptr;
    u32x4 g0;
    g0[0] = 1u;                                              // count=1 (valid)
    g0[1] = lds_addr;                                        // LDS byte address
    g0[2] = (unsigned)(ga & 0xffffffffull);                  // global_addr[31:0]
    g0[3] = (unsigned)((ga >> 32) & 0x1ffffffull) | (2u << 30); // [56:32] | type=2
    i32x8 g1;
    g1[0] = (int)(dsize_code << 16);                         // wg_mask=0, data_size
    g1[1] = (int)(tensor_d0 << 16);                          // tensor_dim0[15:0]
    g1[2] = (int)((tensor_d0 >> 16) | (tensor_d1 << 16));    // dim0 hi | dim1 lo
    g1[3] = (int)((tensor_d1 >> 16) | (tile_d0 << 16));      // dim1 hi | tile_dim0
    g1[4] = (int)(tile_d1 & 0xffffu);                        // tile_dim1, tile_dim2=0
    g1[5] = (int)stride_d0;                                  // tensor_dim0_stride
    g1[6] = 0;
    g1[7] = 0;
    i32x4 z4 = {0, 0, 0, 0};
#if __clang_major__ >= 23
    i32x8 z8 = {0, 0, 0, 0, 0, 0, 0, 0};
    __builtin_amdgcn_tensor_load_to_lds(g0, g1, z4, z4, z8, 0);
#else
    __builtin_amdgcn_tensor_load_to_lds(g0, g1, z4, z4, 0);
#endif
}

// ---------------------------------------------------------------------------
// 1) per-row LayerNorm statistics (mean, inverse stddev)
// ---------------------------------------------------------------------------
__global__ __launch_bounds__(256) void ln_stats(const float* __restrict__ x,
                                                float* __restrict__ mu,
                                                float* __restrict__ rstd) {
    __shared__ float s_sum[256];
    __shared__ float s_sq[256];
    const int row = blockIdx.x;
    const float* xr = x + (size_t)row * HW;
    float s = 0.f, q = 0.f;
    for (int i = threadIdx.x; i < HW; i += 256) {
        float v = xr[i];
        s += v;
        q += v * v;
    }
    s_sum[threadIdx.x] = s;
    s_sq[threadIdx.x]  = q;
    __syncthreads();
    for (int off = 128; off > 0; off >>= 1) {
        if (threadIdx.x < off) {
            s_sum[threadIdx.x] += s_sum[threadIdx.x + off];
            s_sq[threadIdx.x]  += s_sq[threadIdx.x + off];
        }
        __syncthreads();
    }
    if (threadIdx.x == 0) {
        float m = s_sum[0] * (1.f / HW);
        float v = s_sq[0] * (1.f / HW) - m * m;
        mu[row]   = m;
        rstd[row] = rsqrtf(v + LN_EPS);
    }
}

// ---------------------------------------------------------------------------
// 2) weight conversions: w0 -> bf16 ; w1 -> bf16, transposed, padded 49->64
// ---------------------------------------------------------------------------
__global__ __launch_bounds__(256) void cvt_w0(const float* __restrict__ w0,
                                              __bf16* __restrict__ w0b, int n) {
    int i = blockIdx.x * 256 + threadIdx.x;
    if (i < n) w0b[i] = (__bf16)w0[i];
}

__global__ __launch_bounds__(256) void cvt_w1t(const float* __restrict__ w1,
                                               __bf16* __restrict__ w1t) {
    int i = blockIdx.x * 256 + threadIdx.x;   // 64*512 elements
    int n = i >> 9;                            // padded output column
    int k = i & 511;
    float v = (n < KK) ? w1[k * KK + n] : 0.f;
    w1t[i] = (__bf16)v;
}

// ---------------------------------------------------------------------------
// 3) fused LayerNorm -> GEMM1 (bf16 WMMA, fp32 accum) -> exact GELU -> bf16
//    block: 128(M) x 128(N) tile, 8 waves of 16 rows, K-steps of 32 via LDS
// ---------------------------------------------------------------------------
__global__ __launch_bounds__(256) void gemm1_gelu(
        const float* __restrict__ x, const float* __restrict__ mu,
        const float* __restrict__ rstd, const float* __restrict__ gamma,
        const float* __restrict__ beta, const __bf16* __restrict__ Bw,
        const float* __restrict__ b0, __bf16* __restrict__ Hout) {
    __shared__ __bf16 As[128][32];   // [m][k] row-major
    __shared__ __bf16 Bs[128][32];   // [n][k] n-major (so B frag = contiguous)

    const int m0   = blockIdx.y * 128;
    const int n0   = blockIdx.x * 128;
    const int tid  = threadIdx.x;
    const int lane = tid & 31;
    const int wid  = tid >> 5;

    v8f acc[8] = {};

    const int mrow = wid * 16 + (lane & 15);
    const int ka   = (lane >> 4) * 8;
    const int nlo  = lane & 15;
    const int kb2  = (lane >> 4) * 8;

    for (int k0 = 0; k0 < HW; k0 += 32) {
        // stage A: vectorized float4 loads, normalize on the fly, fp32 -> bf16
#pragma unroll
        for (int j = 0; j < 4; ++j) {
            int s = tid + j * 256;            // 1024 slots of 4 floats
            int r = s >> 3, g = s & 7;
            int gr = m0 + r, gk = k0 + g * 4;
            float4 v4 = *(const float4*)(x + (size_t)gr * HW + gk);
            float4 gm = *(const float4*)(gamma + gk);
            float4 bt = *(const float4*)(beta + gk);
            float mr = mu[gr], rs = rstd[gr];
            As[r][g * 4 + 0] = (__bf16)((v4.x - mr) * rs * gm.x + bt.x);
            As[r][g * 4 + 1] = (__bf16)((v4.y - mr) * rs * gm.y + bt.y);
            As[r][g * 4 + 2] = (__bf16)((v4.z - mr) * rs * gm.z + bt.z);
            As[r][g * 4 + 3] = (__bf16)((v4.w - mr) * rs * gm.w + bt.w);
        }
        // stage B transposed: Bs[n][k] from w0b[k*512 + n] (coalesced across lanes)
        {
            int n  = tid & 127;
            int kb = (tid >> 7) * 16;
#pragma unroll
            for (int e = 0; e < 16; ++e)
                Bs[n][kb + e] = Bw[(size_t)(k0 + kb + e) * HID + n0 + n];
        }
        if (k0 + 32 < HW)   // hint next A tile (global_prefetch)
            __builtin_prefetch(x + (size_t)(m0 + (tid >> 1)) * HW + k0 + 32, 0, 0);
        __syncthreads();

        v16bf af = make_frag(*(const v8bf*)&As[mrow][ka],
                             *(const v8bf*)&As[mrow][16 + ka]);
        // grouped B-fragment loads so ds_loads cluster under one wait
#pragma unroll
        for (int g2 = 0; g2 < 2; ++g2) {
            v16bf bfr[4];
#pragma unroll
            for (int t = 0; t < 4; ++t) {
                const v8bf* bp = (const v8bf*)&Bs[(g2 * 4 + t) * 16 + nlo][kb2];
                bfr[t] = make_frag(bp[0], bp[2]);
            }
#pragma unroll
            for (int t = 0; t < 4; ++t)
                acc[g2 * 4 + t] = __builtin_amdgcn_wmma_f32_16x16x32_bf16(
                    false, af, false, bfr[t], (short)0, acc[g2 * 4 + t],
                    false, false);
        }
        __syncthreads();
    }

    // epilogue: + b0, exact GELU, store bf16 hidden
    const int mbase = m0 + wid * 16 + ((lane >> 4) ? 8 : 0);
#pragma unroll
    for (int nt = 0; nt < 8; ++nt) {
        int n = n0 + nt * 16 + nlo;
        float bias = b0[n];
#pragma unroll
        for (int r = 0; r < 8; ++r) {
            float v = acc[nt][r] + bias;
            float g = 0.5f * v * (1.f + erff(v * 0.7071067811865475f));
            Hout[(size_t)(mbase + r) * HID + n] = (__bf16)g;
        }
    }
}

// ---------------------------------------------------------------------------
// 4) GEMM2: hidden(6144x512 bf16) x w1t(64x512 bf16) -> conv kernels fp32
//    B matrix brought in once per block with the Tensor Data Mover
// ---------------------------------------------------------------------------
__global__ __launch_bounds__(256) void gemm2_kern(const __bf16* __restrict__ Hd,
                                                  const __bf16* __restrict__ W1t,
                                                  const float* __restrict__ b1,
                                                  float* __restrict__ Kout) {
    __shared__ __bf16 w1l[NPAD][HID];   // 64 KB, TDM-filled

    const int m0   = blockIdx.x * 128;
    const int lane = threadIdx.x & 31;
    const int wid  = threadIdx.x >> 5;

    if (threadIdx.x < 32) {   // wave 0 issues one TDM DMA for the whole block
        tdm_load_2d((unsigned)(unsigned long long)(void*)&w1l[0][0], W1t,
                    /*tensor_d0=*/HID, /*tensor_d1=*/NPAD,
                    /*tile_d0=*/HID,   /*tile_d1=*/NPAD,
                    /*stride_d0=*/HID, /*dsize=2B*/1u);
        __builtin_amdgcn_s_wait_tensorcnt(0);
    }
    __syncthreads();

    v8f acc[4] = {};

    const int m    = m0 + wid * 16 + (lane & 15);
    const __bf16* arow = Hd + (size_t)m * HID;
    const int ka   = (lane >> 4) * 8;
    const int nlo  = lane & 15;
    const int kb2  = (lane >> 4) * 8;

#pragma unroll 4
    for (int k0 = 0; k0 < HID; k0 += 32) {
        v16bf af = make_frag(*(const v8bf*)(arow + k0 + ka),
                             *(const v8bf*)(arow + k0 + 16 + ka));
#pragma unroll
        for (int nt = 0; nt < 4; ++nt) {
            const v8bf* bp = (const v8bf*)&w1l[nt * 16 + nlo][k0 + kb2];
            v16bf bf = make_frag(bp[0], bp[2]);
            acc[nt] = __builtin_amdgcn_wmma_f32_16x16x32_bf16(
                false, af, false, bf, (short)0, acc[nt], false, false);
        }
    }

    const int mbase = m0 + wid * 16 + ((lane >> 4) ? 8 : 0);
#pragma unroll
    for (int nt = 0; nt < 4; ++nt) {
        int n = nt * 16 + nlo;
        if (n < KK) {
            float bias = b1[n];
#pragma unroll
            for (int r = 0; r < 8; ++r)
                Kout[(size_t)(mbase + r) * KK + n] = acc[nt][r] + bias;
        }
    }
}

// ---------------------------------------------------------------------------
// 5) depthwise 7x7 conv with per-channel predicted kernel + attn_bias
//    one block per (b,c) plane; 62x62 halo tile in LDS; taps in registers
// ---------------------------------------------------------------------------
__global__ __launch_bounds__(256) void dwconv7(const float* __restrict__ x,
                                               const float* __restrict__ Kw,
                                               const float* __restrict__ attn_bias,
                                               float* __restrict__ out) {
    __shared__ float tile[62 * 62];
    __shared__ float taps[KK];
    const int bc = blockIdx.x;
    const float* xp = x + (size_t)bc * HW;

    if (threadIdx.x < KK) taps[threadIdx.x] = Kw[(size_t)bc * KK + threadIdx.x];
    for (int i = threadIdx.x; i < 62 * 62; i += 256) {
        int iy = i / 62 - 3, ix = i % 62 - 3;
        float v = 0.f;
        if (iy >= 0 && iy < HH && ix >= 0 && ix < WW) v = xp[iy * WW + ix];
        tile[i] = v;
    }
    __syncthreads();

    float kw[KK];
#pragma unroll
    for (int t = 0; t < KK; ++t) kw[t] = taps[t];

    const float bias = attn_bias[bc];
    float* op = out + (size_t)bc * HW;
    for (int i = threadIdx.x; i < HW; i += 256) {
        int oy = i / WW, ox = i % WW;
        const float* tp = &tile[oy * 62 + ox];
        float a = bias;
#pragma unroll
        for (int ky = 0; ky < 7; ++ky)
#pragma unroll
            for (int kx = 0; kx < 7; ++kx)
                a = fmaf(tp[ky * 62 + kx], kw[ky * 7 + kx], a);
        op[i] = a;
    }
}

// ---------------------------------------------------------------------------
// launch
// ---------------------------------------------------------------------------
extern "C" void kernel_launch(void* const* d_in, const int* in_sizes, int n_in,
                              void* d_out, int out_size, void* d_ws, size_t ws_size,
                              hipStream_t stream) {
    const float* x     = (const float*)d_in[0];
    const float* gamma = (const float*)d_in[1];
    const float* beta  = (const float*)d_in[2];
    const float* w0    = (const float*)d_in[3];
    const float* b0    = (const float*)d_in[4];
    const float* w1    = (const float*)d_in[5];
    const float* b1    = (const float*)d_in[6];
    const float* abias = (const float*)d_in[7];
    float* out = (float*)d_out;

    // workspace layout (256B-aligned offsets), ~10.8 MB total
    char* ws = (char*)d_ws;
    float*  mu   = (float*)(ws + 0);                 //  6144 f32
    float*  rstd = (float*)(ws + 24576);             //  6144 f32
    __bf16* w0b  = (__bf16*)(ws + 49152);            //  3136*512 bf16
    __bf16* Hd   = (__bf16*)(ws + 3260416);          //  6144*512 bf16
    __bf16* w1t  = (__bf16*)(ws + 9551872);          //  64*512 bf16
    float*  Kw   = (float*)(ws + 9617408);           //  6144*49 f32

    cvt_w0<<<(HW * HID + 255) / 256, 256, 0, stream>>>(w0, w0b, HW * HID);
    cvt_w1t<<<(NPAD * HID) / 256, 256, 0, stream>>>(w1, w1t);
    ln_stats<<<ROWS, 256, 0, stream>>>(x, mu, rstd);
    gemm1_gelu<<<dim3(HID / 128, ROWS / 128), 256, 0, stream>>>(
        x, mu, rstd, gamma, beta, w0b, b0, Hd);
    gemm2_kern<<<ROWS / 128, 256, 0, stream>>>(Hd, w1t, b1, Kw);
    dwconv7<<<ROWS, 256, 0, stream>>>(x, Kw, abias, out);
}
